// Mamba_Encoder_Layer_2302102470701
// MI455X (gfx1250) — matrix operand order repaired and checked by reading, not compiled, not run on hardware
//
#include <hip/hip_runtime.h>
#include <hip/hip_bf16.h>
#include <math.h>

// ---------------- problem constants ----------------
#define D_MODEL  512
#define D_INNER  1024
#define D_STATE  16
#define D_CONV   4
#define DT_RANK  32
#define D_FF     2048
#define BATCH    4
#define SEQ      2048
#define BL       (BATCH * SEQ)   // 8192 rows

#define KC       128             // k-chunk staged per TDM transfer
#define NTILE    64              // output columns per block
#define MT       2               // 16-row M tiles per wave

typedef __bf16 bf16_t;
typedef bf16_t v16bf __attribute__((ext_vector_type(16)));
typedef bf16_t v8bf  __attribute__((ext_vector_type(8)));
typedef float  v8f   __attribute__((ext_vector_type(8)));
typedef unsigned int v4u __attribute__((ext_vector_type(4)));
typedef int    v8i_  __attribute__((ext_vector_type(8)));
typedef int    v4i_  __attribute__((ext_vector_type(4)));

union Frag16 {
    v16bf v;
    v8bf  h[2];
};

enum { ACT_NONE = 0, ACT_GELU = 1 };

// ---------------- fp32 -> bf16 convert ----------------
__global__ void cvt_bf16_kernel(const float* __restrict__ in, bf16_t* __restrict__ out, size_t n) {
    size_t i = (size_t)blockIdx.x * blockDim.x + threadIdx.x;
    size_t stride = (size_t)gridDim.x * blockDim.x;
    for (; i < n; i += stride) out[i] = (bf16_t)in[i];
}

// ---------------- TDM: DMA one (rows x KC) bf16 tile (row stride K elems) into LDS ----------------
// D# layout per CDNA5 ISA ch.8: group0 = {count/lds_addr/global_addr/type}, group1 = dims/strides.
__device__ __forceinline__ void tdm_load_b_tile(const bf16_t* gsrc, unsigned int lds_off,
                                                int rows, int K) {
    unsigned long long ga = (unsigned long long)(uintptr_t)gsrc;
    v4u g0;
    g0[0] = 1u;                                         // count=1, user mode, no gather
    g0[1] = lds_off;                                    // lds_addr (bytes)
    g0[2] = (unsigned int)ga;                           // global_addr[31:0]
    g0[3] = (unsigned int)((ga >> 32) & 0x1FFFFFFu)     // global_addr[56:32]
          | 0x80000000u;                                // type=2 ("image") in bits 127:126
    v8i_ g1;
    g1[0] = 0x00010000;                                 // wg_mask=0, data_size=1 (2 bytes)
    g1[1] = (int)((K & 0xFFFF) << 16);                  // tensor_dim0[15:0] @ bits 63:48
    g1[2] = (int)(((unsigned)K >> 16) & 0xFFFF)         // tensor_dim0[31:16]
          | (int)((rows & 0xFFFF) << 16);               // tensor_dim1[15:0]
    g1[3] = (int)(KC << 16);                            // tensor_dim1[31:16]=0, tile_dim0=KC
    g1[4] = rows;                                       // tile_dim1=rows, tile_dim2=0
    g1[5] = K;                                          // tensor_dim0_stride[31:0]
    g1[6] = 0;                                          // stride0[47:32]=0, stride1 lo=0
    g1[7] = 0;
    v4i_ z4 = {0, 0, 0, 0};
#if defined(__clang_major__) && (__clang_major__ >= 23)
    v8i_ z8 = {0, 0, 0, 0, 0, 0, 0, 0};
    __builtin_amdgcn_tensor_load_to_lds(g0, g1, z4, z4, z8, 0);
#else
    __builtin_amdgcn_tensor_load_to_lds(g0, g1, z4, z4, 0);
#endif
}

// ---------------- WMMA GEMM: C[M,N] = act(A[M,K] @ B[N,K]^T + bias) ----------------
// A: bf16 row-major (lda=K), B: bf16 row-major N x K (ldb=K).
// Block: 256 threads = 8 waves; wave w -> 32 rows [m0, m0+32), all waves share cols [n0, n0+64).
// B panel (64 x KC) staged into LDS by the Tensor Data Mover, double-buffered (TENSORcnt overlap);
// per 32-k step: 4 global_load_b128 (A) + 8 ds_load_b128 (B) feed 8 back-to-back WMMAs.
// Requires: M % 256 == 0, N % 64 == 0, K % KC == 0.
template <typename OutT, int ACT, bool HAS_BIAS>
__global__ __launch_bounds__(256)
void wmma_gemm_kernel(const bf16_t* __restrict__ A, const bf16_t* __restrict__ Bm,
                      const float* __restrict__ bias, OutT* __restrict__ C,
                      int M, int N, int K) {
    __shared__ __align__(16) bf16_t sB[2][NTILE * KC];

    const int lane = threadIdx.x & 31;
    const int wave = threadIdx.x >> 5;
    const int l16  = lane & 15;
    const int hi   = lane >> 4;

    const int m0 = (blockIdx.y * 8 + wave) * (16 * MT);
    const int n0 = blockIdx.x * NTILE;

    v8f acc[MT][4] = {};

    // A fragment sources: lane(l16,hi) holds row m0+u*16+l16, K chunks {hi*8..} and {16+hi*8..}
    const bf16_t* arow[MT];
#pragma unroll
    for (int u = 0; u < MT; ++u)
        arow[u] = A + (size_t)(m0 + u * 16 + l16) * K + hi * 8;

    const int nchunks = K / KC;

    if (wave == 0)
        tdm_load_b_tile(Bm + (size_t)n0 * K, (unsigned int)(uintptr_t)&sB[0][0], NTILE, K);

    for (int c = 0; c < nchunks; ++c) {
        __syncthreads();                 // all waves done reading buffer (c+1)&1 (chunk c-1)
        if (wave == 0) {
            if (c + 1 < nchunks) {
                tdm_load_b_tile(Bm + (size_t)n0 * K + (c + 1) * KC,
                                (unsigned int)(uintptr_t)&sB[(c + 1) & 1][0], NTILE, K);
                __builtin_amdgcn_s_wait_tensorcnt((short)1);   // chunk c has landed
            } else {
                __builtin_amdgcn_s_wait_tensorcnt((short)0);
            }
        }
        __syncthreads();                 // publish chunk c to all waves
        const bf16_t* bbase = &sB[c & 1][0];

#pragma unroll
        for (int kk = 0; kk < KC; kk += 32) {
            const int k0 = c * KC + kk;
            // all A fragments for this k-step
            Frag16 fa[MT];
#pragma unroll
            for (int u = 0; u < MT; ++u) {
                fa[u].h[0] = *(const v8bf*)(arow[u] + k0);        // K = k0 + hi*8 + [0..7]
                fa[u].h[1] = *(const v8bf*)(arow[u] + k0 + 16);   // K = k0 + 16 + hi*8 + [0..7]
            }
            // all B fragments for this k-step (from LDS)
            Frag16 fb[4];
#pragma unroll
            for (int t = 0; t < 4; ++t) {
                const bf16_t* brow = bbase + (t * 16 + l16) * KC + kk + hi * 8;
                fb[t].h[0] = *(const v8bf*)(brow);
                fb[t].h[1] = *(const v8bf*)(brow + 16);
            }
            // 8 back-to-back WMMAs, independent accumulators
#pragma unroll
            for (int u = 0; u < MT; ++u)
#pragma unroll
                for (int t = 0; t < 4; ++t)
                    acc[u][t] = __builtin_amdgcn_wmma_f32_16x16x32_bf16(
                        false, fa[u].v, false, fb[t].v, (short)0, acc[u][t], false, false);
        }
    }

    // epilogue: C layout lane(l16,hi), vgpr r -> (m = m0 + u*16 + hi*8 + r, n = n0 + t*16 + l16)
#pragma unroll
    for (int t = 0; t < 4; ++t) {
        const int n = n0 + t * 16 + l16;
        float bs = 0.0f;
        if (HAS_BIAS) bs = bias[n];
#pragma unroll
        for (int u = 0; u < MT; ++u) {
            const int mrow = m0 + u * 16 + hi * 8;
#pragma unroll
            for (int r = 0; r < 8; ++r) {
                float v = acc[u][t][r] + bs;
                if (ACT == ACT_GELU) v = 0.5f * v * (1.0f + erff(v * 0.70710678118654752f));
                C[(size_t)(mrow + r) * N + n] = (OutT)v;
            }
        }
    }
}

// ---------------- causal depthwise conv (width 4) + SiLU ----------------
// xz_bf: (BL, 2*D_INNER) bf16; xp = cols [0, D_INNER). Output xc_bf: (BL, D_INNER) bf16.
__global__ __launch_bounds__(256)
void conv_silu_kernel(const bf16_t* __restrict__ xz, const float* __restrict__ Wc,
                      const float* __restrict__ bc, bf16_t* __restrict__ xc) {
    size_t idx = (size_t)blockIdx.x * blockDim.x + threadIdx.x;
    if (idx >= (size_t)BL * D_INNER) return;
    const int d = (int)(idx % D_INNER);
    const size_t m = idx / D_INNER;
    const int l = (int)(m % SEQ);

    float accv = bc[d];
#pragma unroll
    for (int j = 0; j < D_CONV; ++j) {
        int li = l - (D_CONV - 1) + j;
        if (li >= 0) {
            float xv = (float)xz[(m - (size_t)(D_CONV - 1 - j)) * (2 * D_INNER) + d];
            accv += xv * Wc[d * D_CONV + j];
        }
    }
    accv = accv / (1.0f + __expf(-accv));   // SiLU
    xc[idx] = (bf16_t)accv;
}

// ---------------- dt projection: softplus(dbl[:, :32] @ W_dt^T + b_dt) ----------------
// grid = (D_INNER/256, BL), block = 256
__global__ __launch_bounds__(256)
void dt_kernel(const float* __restrict__ dbl, const float* __restrict__ Wdt,
               const float* __restrict__ bdt, float* __restrict__ dt) {
    __shared__ float s[DT_RANK];
    const size_t m = blockIdx.y;
    const int d = blockIdx.x * 256 + threadIdx.x;
    if (threadIdx.x < DT_RANK) s[threadIdx.x] = dbl[m * (DT_RANK + 2 * D_STATE) + threadIdx.x];
    __syncthreads();

    float a = bdt[d];
#pragma unroll
    for (int j = 0; j < DT_RANK; ++j) a += s[j] * Wdt[d * DT_RANK + j];
    float sp = (a > 20.0f) ? a : log1pf(__expf(a));     // stable softplus
    dt[m * D_INNER + d] = sp;
}

// ---------------- selective scan + gating ----------------
// One thread per (b, d) channel; 16-state recurrence in registers.
// B/C (32 floats per timestep) double-buffered in LDS -> one barrier per step.
// Fuses y = (scan + xc*D) * silu(z), writes bf16 y.
// grid = B * D_INNER / 256 = 16 blocks, block = 256
__global__ __launch_bounds__(256)
void scan_kernel(const float* __restrict__ dt, const bf16_t* __restrict__ xc,
                 const float* __restrict__ dbl, const float* __restrict__ A_log,
                 const float* __restrict__ Dvec, const bf16_t* __restrict__ xz,
                 bf16_t* __restrict__ y) {
    const int b = blockIdx.x >> 2;                       // 4 blocks per batch
    const int d = ((blockIdx.x & 3) << 8) + threadIdx.x; // channel

    float Ar[D_STATE];
#pragma unroll
    for (int n = 0; n < D_STATE; ++n) Ar[n] = -__expf(A_log[d * D_STATE + n]);
    float h[D_STATE];
#pragma unroll
    for (int n = 0; n < D_STATE; ++n) h[n] = 0.0f;
    const float Dd = Dvec[d];

    __shared__ float sBC[2][2 * D_STATE];
    const size_t mbase = (size_t)b * SEQ;

    if (threadIdx.x < 2 * D_STATE)
        sBC[0][threadIdx.x] = dbl[mbase * (DT_RANK + 2 * D_STATE) + DT_RANK + threadIdx.x];
    __syncthreads();

    for (int l = 0; l < SEQ; ++l) {
        const size_t m = mbase + l;
        const int p = l & 1;
        if (threadIdx.x < 2 * D_STATE && (l + 1) < SEQ)
            sBC[p ^ 1][threadIdx.x] =
                dbl[(m + 1) * (DT_RANK + 2 * D_STATE) + DT_RANK + threadIdx.x];

        const float dtv = dt[m * D_INNER + d];
        const float xcv = (float)xc[m * D_INNER + d];
        const float dbx = dtv * xcv;
        float yv = 0.0f;
#pragma unroll
        for (int n = 0; n < D_STATE; ++n) {
            float dA = __expf(dtv * Ar[n]);
            h[n] = dA * h[n] + dbx * sBC[p][n];
            yv += h[n] * sBC[p][D_STATE + n];
        }
        yv += xcv * Dd;
        const float zv = (float)xz[m * (2 * D_INNER) + D_INNER + d];
        yv *= zv / (1.0f + __expf(-zv));
        y[m * D_INNER + d] = (bf16_t)yv;

        __syncthreads();   // next-step stage written & this-step reads done
    }
}

// ---------------- host orchestration ----------------
static inline int cvt_blocks(size_t n) {
    size_t b = (n + 255) / 256;
    return (int)(b > 4096 ? 4096 : b);
}

extern "C" void kernel_launch(void* const* d_in, const int* in_sizes, int n_in,
                              void* d_out, int out_size, void* d_ws, size_t ws_size,
                              hipStream_t stream) {
    (void)in_sizes; (void)n_in; (void)out_size; (void)ws_size;

    const float* x      = (const float*)d_in[0];   // (B,L,512)
    const float* W_in   = (const float*)d_in[1];   // (2048,512)
    const float* W_conv = (const float*)d_in[2];   // (1024,1,4)
    const float* b_conv = (const float*)d_in[3];   // (1024)
    const float* W_xprj = (const float*)d_in[4];   // (64,1024)
    const float* W_dt   = (const float*)d_in[5];   // (1024,32)
    const float* b_dt   = (const float*)d_in[6];   // (1024)
    const float* A_log  = (const float*)d_in[7];   // (1024,16)
    const float* Dvec   = (const float*)d_in[8];   // (1024)
    const float* W_out  = (const float*)d_in[9];   // (512,1024)
    const float* W1     = (const float*)d_in[10];  // (2048,512)
    const float* b1     = (const float*)d_in[11];  // (2048)
    const float* W2     = (const float*)d_in[12];  // (512,2048)
    const float* b2     = (const float*)d_in[13];  // (512)
    float* out          = (float*)d_out;           // (B,L,512)

    // workspace carve (256B aligned)
    char* p = (char*)d_ws;
    auto alloc = [&](size_t bytes) -> char* {
        char* r = p;
        p += (bytes + 255) & ~(size_t)255;
        return r;
    };
    bf16_t* x_bf     = (bf16_t*)alloc((size_t)BL * D_MODEL * 2);
    bf16_t* w_in_bf  = (bf16_t*)alloc((size_t)(2 * D_INNER) * D_MODEL * 2);
    bf16_t* w_xp_bf  = (bf16_t*)alloc((size_t)(DT_RANK + 2 * D_STATE) * D_INNER * 2);
    bf16_t* w_out_bf = (bf16_t*)alloc((size_t)D_MODEL * D_INNER * 2);
    bf16_t* w1_bf    = (bf16_t*)alloc((size_t)D_FF * D_MODEL * 2);
    bf16_t* w2_bf    = (bf16_t*)alloc((size_t)D_MODEL * D_FF * 2);
    bf16_t* xz_bf    = (bf16_t*)alloc((size_t)BL * 2 * D_INNER * 2);
    bf16_t* xc_bf    = (bf16_t*)alloc((size_t)BL * D_INNER * 2);
    float*  dbl      = (float*)alloc((size_t)BL * (DT_RANK + 2 * D_STATE) * 4);
    float*  dt       = (float*)alloc((size_t)BL * D_INNER * 4);
    bf16_t* y_bf     = (bf16_t*)alloc((size_t)BL * D_INNER * 2);
    bf16_t* h_bf     = (bf16_t*)alloc((size_t)BL * D_MODEL * 2);
    bf16_t* f1_bf    = (bf16_t*)alloc((size_t)BL * D_FF * 2);

    // 1) converts
    {
        size_t n;
        n = (size_t)BL * D_MODEL;
        cvt_bf16_kernel<<<cvt_blocks(n), 256, 0, stream>>>(x, x_bf, n);
        n = (size_t)(2 * D_INNER) * D_MODEL;
        cvt_bf16_kernel<<<cvt_blocks(n), 256, 0, stream>>>(W_in, w_in_bf, n);
        n = (size_t)(DT_RANK + 2 * D_STATE) * D_INNER;
        cvt_bf16_kernel<<<cvt_blocks(n), 256, 0, stream>>>(W_xprj, w_xp_bf, n);
        n = (size_t)D_MODEL * D_INNER;
        cvt_bf16_kernel<<<cvt_blocks(n), 256, 0, stream>>>(W_out, w_out_bf, n);
        n = (size_t)D_FF * D_MODEL;
        cvt_bf16_kernel<<<cvt_blocks(n), 256, 0, stream>>>(W1, w1_bf, n);
        n = (size_t)D_MODEL * D_FF;
        cvt_bf16_kernel<<<cvt_blocks(n), 256, 0, stream>>>(W2, w2_bf, n);
    }

    // 2) in_proj: xz = x @ W_in^T   (M=8192, N=2048, K=512) -> bf16
    wmma_gemm_kernel<bf16_t, ACT_NONE, false>
        <<<dim3((2 * D_INNER) / NTILE, BL / 256), 256, 0, stream>>>(
            x_bf, w_in_bf, nullptr, xz_bf, BL, 2 * D_INNER, D_MODEL);

    // 3) causal depthwise conv + SiLU -> xc (bf16)
    conv_silu_kernel<<<((size_t)BL * D_INNER) / 256, 256, 0, stream>>>(
        xz_bf, W_conv, b_conv, xc_bf);

    // 4) x_proj: dbl = xc @ W_xproj^T   (M=8192, N=64, K=1024) -> fp32
    wmma_gemm_kernel<float, ACT_NONE, false>
        <<<dim3((DT_RANK + 2 * D_STATE) / NTILE, BL / 256), 256, 0, stream>>>(
            xc_bf, w_xp_bf, nullptr, dbl, BL, DT_RANK + 2 * D_STATE, D_INNER);

    // 5) dt = softplus(dbl[:, :32] @ W_dt^T + b_dt)  (K=32: VALU)
    dt_kernel<<<dim3(D_INNER / 256, BL), 256, 0, stream>>>(dbl, W_dt, b_dt, dt);

    // 6) selective scan + (+xc*D) + silu(z) gating -> y (bf16)
    scan_kernel<<<(BATCH * D_INNER) / 256, 256, 0, stream>>>(
        dt, xc_bf, dbl, A_log, Dvec, xz_bf, y_bf);

    // 7) out_proj: h = y @ W_out^T   (M=8192, N=512, K=1024) -> bf16
    wmma_gemm_kernel<bf16_t, ACT_NONE, false>
        <<<dim3(D_MODEL / NTILE, BL / 256), 256, 0, stream>>>(
            y_bf, w_out_bf, nullptr, h_bf, BL, D_MODEL, D_INNER);

    // 8) FFN1: f1 = gelu(h @ W1^T + b1)   (M=8192, N=2048, K=512) -> bf16
    wmma_gemm_kernel<bf16_t, ACT_GELU, true>
        <<<dim3(D_FF / NTILE, BL / 256), 256, 0, stream>>>(
            h_bf, w1_bf, b1, f1_bf, BL, D_FF, D_MODEL);

    // 9) FFN2: out = f1 @ W2^T + b2   (M=8192, N=512, K=2048) -> fp32 d_out
    wmma_gemm_kernel<float, ACT_NONE, true>
        <<<dim3(D_MODEL / NTILE, BL / 256), 256, 0, stream>>>(
            f1_bf, w2_bf, b2, out, BL, D_MODEL, D_FF);
}
